// C_Attention_55473797595256
// MI455X (gfx1250) — matrix operand order repaired and checked by reading, not compiled, not run on hardware
//
#include <hip/hip_runtime.h>
#include <hip/hip_bf16.h>

// ---------------------------------------------------------------------------
// XCA channel attention for MI455X (gfx1250, wave32, WMMA).
// Pixel-major activation layout [N][C]: WMMA A-fragments (pixels x K) and
// B-fragments (weight rows) are both K-contiguous -> pure global->VGPR->WMMA
// pipeline with no LDS staging in the heavy GEMMs. bf16 in, f32 accumulate.
// Wave tile = 32 px x 64 och (2 A-frags x 4 B-frags -> 8 WMMA per K-step).
// ---------------------------------------------------------------------------

typedef __attribute__((ext_vector_type(16))) __bf16 v16bf;
typedef __attribute__((ext_vector_type(8)))  __bf16 v8bf;
typedef __attribute__((ext_vector_type(8)))  float  v8f;

#define NS 65536   // H*W
#define HH 256
#define WW 256
#define CB 192
#define NB 2

static __device__ __forceinline__ v16bf pack16(v8bf lo, v8bf hi) {
  v16bf r;
#pragma unroll
  for (int i = 0; i < 8; ++i) { r[i] = lo[i]; r[i + 8] = hi[i]; }
  return r;
}

// A fragment (16x32 bf16, M x K), rows K-contiguous in memory.
// ISA layout: lanes 0-15 (M=lane) hold K {0..7, 16..23};
//             lanes 16-31 (M=lane-16) hold K {8..15, 24..31}.
static __device__ __forceinline__ v16bf load_a_frag(const __bf16* __restrict__ A,
                                                    long ldk, int mBase, int kb,
                                                    int lm, int half) {
  const __bf16* row = A + (long)(mBase + lm) * ldk + kb;
  v8bf a0 = *(const v8bf*)(row + half * 8);
  v8bf a1 = *(const v8bf*)(row + 16 + half * 8);
  return pack16(a0, a1);
}

// B fragment (32x16 bf16, K x N) where column n's K-values are contiguous
// (row n of the row-major weight matrix). Lanes 0-15 (N=lane): K {0..7, 16..23};
// lanes 16-31: K {8..15, 24..31}.
static __device__ __forceinline__ v16bf load_b_frag_row(const __bf16* __restrict__ row,
                                                        int kb, int half) {
  v8bf b0 = *(const v8bf*)(row + kb + half * 8);
  v8bf b1 = *(const v8bf*)(row + kb + 16 + half * 8);
  return pack16(b0, b1);
}

// ---------------------------------------------------------------------------
// Pixel-major GEMM: Y[p][m] = sum_k Xp[p][k] * W[m][k].
// Block = 128 threads (4 waves); block tile 128 px x 64 och; wave tile
// 32 px x 64 och. No LDS in the K loop. TOUT_CM=true stores the f32 result
// channel-major (transposed through LDS) for the final NCHW output.
// ---------------------------------------------------------------------------
template <typename TOUT, bool TOUT_CM>
__global__ __launch_bounds__(128)
void gemm_px_bf16(const __bf16* __restrict__ W, const __bf16* __restrict__ Xp,
                  TOUT* __restrict__ D, int M, int K, long ldOut,
                  long bsW, long bsX, long bsD) {
  const int b = blockIdx.z;
  W += (long)b * bsW; Xp += (long)b * bsX; D += (long)b * bsD;
  const int pBase = blockIdx.x * 128;
  const int oBase = blockIdx.y * 64;
  const int tid  = threadIdx.x;
  const int wave = tid >> 5;
  const int lane = tid & 31;
  const int lm = lane & 15, half = lane >> 4;
  const int pWave = pBase + wave * 32;

  const __bf16* wrow[4];
#pragma unroll
  for (int j = 0; j < 4; ++j) wrow[j] = W + (long)(oBase + j * 16 + lm) * K;
  const __bf16* arow0 = Xp + (long)(pWave + lm) * K;
  const __bf16* arow1 = Xp + (long)(pWave + 16 + lm) * K;

  v8f acc[2][4] = {};
  for (int kb = 0; kb < K; kb += 32) {
    __builtin_prefetch(arow0 + kb + 32, 0, 1);     // global_prefetch_b8
    __builtin_prefetch(arow1 + kb + 32, 0, 1);
    v16bf af0 = load_a_frag(Xp, K, pWave,      kb, lm, half);
    v16bf af1 = load_a_frag(Xp, K, pWave + 16, kb, lm, half);
#pragma unroll
    for (int j = 0; j < 4; ++j) {
      v16bf bfr = load_b_frag_row(wrow[j], kb, half);
      acc[0][j] = __builtin_amdgcn_wmma_f32_16x16x32_bf16(
          false, af0, false, bfr, (short)0, acc[0][j], false, false);
      acc[1][j] = __builtin_amdgcn_wmma_f32_16x16x32_bf16(
          false, af1, false, bfr, (short)0, acc[1][j], false, false);
    }
  }

  if constexpr (!TOUT_CM) {
    // pixel-major store: D[p][och]
#pragma unroll
    for (int s = 0; s < 2; ++s)
#pragma unroll
      for (int j = 0; j < 4; ++j)
#pragma unroll
        for (int r = 0; r < 8; ++r) {
          int p = pWave + s * 16 + r + 8 * half;
          int o = oBase + j * 16 + lm;
          D[(long)p * M + o] = (TOUT)acc[s][j][r];
        }
  } else {
    // channel-major f32 store via LDS transpose (coalesced NCHW rows)
    __shared__ float t[64 * 130];
#pragma unroll
    for (int s = 0; s < 2; ++s)
#pragma unroll
      for (int j = 0; j < 4; ++j)
#pragma unroll
        for (int r = 0; r < 8; ++r) {
          int pl = wave * 32 + s * 16 + r + 8 * half;
          int ol = j * 16 + lm;
          t[ol * 130 + pl] = acc[s][j][r];
        }
    __syncthreads();
#pragma unroll
    for (int i = 0; i < 64; ++i) {
      int idx = tid + i * 128;           // 8192 elements
      int o = idx >> 7, p = idx & 127;
      D[(long)(oBase + o) * ldOut + pBase + p] = t[o * 130 + p];
    }
  }
}

// ---------------------------------------------------------------------------
// Full 3x3 conv (groups=1) as implicit-im2col GEMM, pixel-major.
// M=192, K=9*192. A 128-pixel tile lies inside one image row, so each tap
// needs one uniform y-check plus per-lane x-checks (branchless selects).
// ---------------------------------------------------------------------------
__global__ __launch_bounds__(128)
void conv3_px_bf16(const __bf16* __restrict__ W3, const __bf16* __restrict__ Xp,
                   __bf16* __restrict__ D, long bsX, long bsD) {
  const int b = blockIdx.z;
  Xp += (long)b * bsX; D += (long)b * bsD;
  const int pBase = blockIdx.x * 128;
  const int oBase = blockIdx.y * 64;
  const int tid  = threadIdx.x;
  const int wave = tid >> 5;
  const int lane = tid & 31;
  const int lm = lane & 15, half = lane >> 4;
  const int pWave = pBase + wave * 32;
  const int yRow  = pBase >> 8;          // uniform image row of this tile
  const int xLane = (pWave & 255) + lm;  // lane's x coordinate (subtile 0)

  const __bf16* wrow[4];
#pragma unroll
  for (int j = 0; j < 4; ++j) wrow[j] = W3 + (long)(oBase + j * 16 + lm) * (9 * CB);

  v8f acc[2][4] = {};
  const v8bf vzero = {};
  for (int t = 0; t < 9; ++t) {
    const int dy = t / 3 - 1, dx = t % 3 - 1;    // scalar (t is uniform)
    const int yy = yRow + dy;
    const bool yok = (unsigned)yy < (unsigned)HH;
    const int xs0 = xLane + dx;
    const int xs1 = xLane + 16 + dx;
    const bool ok0 = yok && ((unsigned)xs0 < (unsigned)WW);
    const bool ok1 = yok && ((unsigned)xs1 < (unsigned)WW);
    const long prow0 = ok0 ? ((long)((yy << 8) + xs0)) : 0L;
    const long prow1 = ok1 ? ((long)((yy << 8) + xs1)) : 0L;
    const __bf16* arow0 = Xp + prow0 * CB;
    const __bf16* arow1 = Xp + prow1 * CB;
#pragma unroll
    for (int kc = 0; kc < CB; kc += 32) {
      v8bf a00 = *(const v8bf*)(arow0 + kc + half * 8);
      v8bf a01 = *(const v8bf*)(arow0 + kc + 16 + half * 8);
      v8bf a10 = *(const v8bf*)(arow1 + kc + half * 8);
      v8bf a11 = *(const v8bf*)(arow1 + kc + 16 + half * 8);
      a00 = ok0 ? a00 : vzero;  a01 = ok0 ? a01 : vzero;
      a10 = ok1 ? a10 : vzero;  a11 = ok1 ? a11 : vzero;
      v16bf af0 = pack16(a00, a01);
      v16bf af1 = pack16(a10, a11);
      const int kg = t * CB + kc;
#pragma unroll
      for (int j = 0; j < 4; ++j) {
        v16bf bfr = load_b_frag_row(wrow[j], kg, half);
        acc[0][j] = __builtin_amdgcn_wmma_f32_16x16x32_bf16(
            false, af0, false, bfr, (short)0, acc[0][j], false, false);
        acc[1][j] = __builtin_amdgcn_wmma_f32_16x16x32_bf16(
            false, af1, false, bfr, (short)0, acc[1][j], false, false);
      }
    }
  }
#pragma unroll
  for (int s = 0; s < 2; ++s)
#pragma unroll
    for (int j = 0; j < 4; ++j)
#pragma unroll
      for (int r = 0; r < 8; ++r) {
        int p = pWave + s * 16 + r + 8 * half;
        int o = oBase + j * 16 + lm;
        D[(long)p * CB + o] = (__bf16)acc[s][j][r];
      }
}

// ---------------------------------------------------------------------------
// Gram: per (b,h) G[48,48] += q_h^T * k_h over pixels (K=65536), pixel chunks
// split over 32 blocks of 128 threads. Pixel-major q/k tiles are transposed
// into LDS so both fragments read contiguous K (pixels).
// ---------------------------------------------------------------------------
__global__ __launch_bounds__(128)
void gram48_px(const __bf16* __restrict__ Q, const __bf16* __restrict__ Km,
               float* __restrict__ G) {
  __shared__ __attribute__((aligned(16))) __bf16 qt[48 * 64];
  __shared__ __attribute__((aligned(16))) __bf16 kt[48 * 64];
  const int bh = blockIdx.y;                 // b*4+h
  const int b = bh >> 2, h = bh & 3;
  const __bf16* q = Q  + (long)b * NS * CB + h * 48;
  const __bf16* k = Km + (long)b * NS * CB + h * 48;
  float* g = G + (long)bh * 48 * 48;
  const int tid = threadIdx.x;
  const int wave = tid >> 5;
  const int lane = tid & 31;
  const int lm = lane & 15, half = lane >> 4;
  const long p0 = (long)blockIdx.x * (NS / 32);

  // wave w owns WMMA tiles {w, w+4, w+8} of the 3x3 tile grid
  v8f acc[3] = {};
  for (int sub = 0; sub < 32; ++sub) {
    const long ps = p0 + sub * 64;
    __syncthreads();
#pragma unroll
    for (int i = 0; i < 24; ++i) {       // 48*64 = 3072 elems per matrix
      int idx = tid + i * 128;
      int p = idx / 48, c = idx % 48;    // consecutive tid -> c (coalesced)
      qt[c * 64 + p] = q[(ps + p) * CB + c];
      kt[c * 64 + p] = k[(ps + p) * CB + c];
    }
    __syncthreads();
#pragma unroll
    for (int ti = 0; ti < 3; ++ti) {
      int tile = wave + ti * 4;
      if (tile >= 9) break;
      int r = tile / 3, c = tile % 3;
#pragma unroll
      for (int kb = 0; kb < 64; kb += 32) {
        v16bf af  = load_a_frag(qt, 64, r * 16, kb, lm, half);
        v16bf bfr = load_b_frag_row(kt + (long)(c * 16 + lm) * 64, kb, half);
        acc[ti] = __builtin_amdgcn_wmma_f32_16x16x32_bf16(
            false, af, false, bfr, (short)0, acc[ti], false, false);
      }
    }
  }
#pragma unroll
  for (int ti = 0; ti < 3; ++ti) {
    int tile = wave + ti * 4;
    if (tile >= 9) break;
    int r = tile / 3, c = tile % 3;
#pragma unroll
    for (int j = 0; j < 8; ++j)
      atomicAdd(&g[(r * 16 + j + 8 * half) * 48 + c * 16 + lm], acc[ti][j]);
  }
}

// ---------------------------------------------------------------------------
// Small helper kernels
// ---------------------------------------------------------------------------

// fp32 [C][N] -> bf16 [N][C] tiled transpose-cast (both sides coalesced)
__global__ __launch_bounds__(256)
void tcast_k(const float* __restrict__ src, __bf16* __restrict__ dst) {
  __shared__ float t[32 * 65];
  const long bo = (long)blockIdx.z * CB * NS;
  const int p0 = blockIdx.x * 64, c0 = blockIdx.y * 32;
  const int tid = threadIdx.x;
#pragma unroll
  for (int i = 0; i < 8; ++i) {          // 32x64 = 2048 elements
    int idx = tid + i * 256;
    int c = idx >> 6, p = idx & 63;
    t[c * 65 + p] = src[bo + (long)(c0 + c) * NS + p0 + p];
  }
  __syncthreads();
#pragma unroll
  for (int i = 0; i < 8; ++i) {
    int idx = tid + i * 256;
    int p = idx >> 5, c = idx & 31;
    dst[bo + (long)(p0 + p) * CB + c0 + c] = (__bf16)t[c * 65 + p];
  }
}

__global__ void cast_f32_bf16_k(const float* __restrict__ s, __bf16* __restrict__ d, long n) {
  long i = (long)blockIdx.x * blockDim.x + threadIdx.x;
  if (i < n) d[i] = (__bf16)s[i];
}

__global__ void zero_f32_k(float* __restrict__ p, long n) {
  long i = (long)blockIdx.x * blockDim.x + threadIdx.x;
  if (i < n) p[i] = 0.0f;
}

// pack v_dw_w [O=192][I=192][3][3] fp32 -> bf16 [O][tap*192+i]
__global__ void pack_vdw_k(const float* __restrict__ w, __bf16* __restrict__ out) {
  long i = (long)blockIdx.x * blockDim.x + threadIdx.x;
  if (i >= (long)CB * 9 * CB) return;
  int o = (int)(i / (9 * CB));
  int kk = (int)(i % (9 * CB));
  int t = kk / CB, ci = kk % CB;
  out[i] = (__bf16)w[((long)o * CB + ci) * 9 + t];
}

// depthwise 3x3 (groups=384) on pixel-major qk[N][384]; split into q/k [N][192]
// block = 384 threads (one pixel per block, thread = channel -> coalesced)
__global__ __launch_bounds__(384)
void dwconv3_px_k(const __bf16* __restrict__ qk, const float* __restrict__ wdw,
                  __bf16* __restrict__ qout, __bf16* __restrict__ kout) {
  const int p = blockIdx.x;
  const int b = blockIdx.y;
  const int c = threadIdx.x;
  const int y = p >> 8, x = p & 255;
  const __bf16* base = qk + (long)b * NS * 2 * CB;
  const float* w = wdw + c * 9;
  float acc = 0.0f;
#pragma unroll
  for (int dy = -1; dy <= 1; ++dy)
#pragma unroll
    for (int dx = -1; dx <= 1; ++dx) {
      int yy = y + dy, xx = x + dx;
      if ((unsigned)yy < (unsigned)HH && (unsigned)xx < (unsigned)WW)
        acc += w[(dy + 1) * 3 + (dx + 1)] *
               (float)base[(long)((yy << 8) + xx) * (2 * CB) + c];
    }
  __bf16 r = (__bf16)acc;
  long po = (long)b * NS * CB + (long)p * CB;
  if (c < CB) qout[po + c] = r;
  else        kout[po + (c - CB)] = r;
}

// column (=channel) sum of squares over a pixel chunk, atomic into acc[b*192+c]
__global__ __launch_bounds__(192)
void colsq_k(const __bf16* __restrict__ Q, float* __restrict__ acc) {
  const int c = threadIdx.x;
  const int b = blockIdx.y;
  const long p0 = (long)blockIdx.x * 256;
  const __bf16* base = Q + (long)b * NS * CB;
  float s = 0.0f;
  for (int i = 0; i < 256; ++i) {
    float v = (float)base[(p0 + i) * CB + c];
    s += v * v;
  }
  atomicAdd(&acc[b * CB + c], s);
}

__global__ void final_norm_k(float* __restrict__ p, int n) {
  int i = blockIdx.x * blockDim.x + threadIdx.x;
  if (i < n) p[i] = fmaxf(sqrtf(p[i]), 1e-12f);
}

// softmax over d of G[b,h,c,d] * temp[h] / (|q_c| |k_d|)
__global__ void softmax48_k(const float* __restrict__ G, const float* __restrict__ qn,
                            const float* __restrict__ kn, const float* __restrict__ temp,
                            float* __restrict__ attn) {
  int idx = blockIdx.x * blockDim.x + threadIdx.x;
  if (idx >= NB * 4 * 48) return;
  int c = idx % 48, bh = idx / 48, h = bh & 3, b = bh >> 2;
  const float* grow = G + (long)bh * 2304 + c * 48;
  float t = temp[h];
  float qv = fmaxf(qn[b * CB + h * 48 + c], 1e-12f);
  float vals[48], mx = -3.0e38f;
  for (int d = 0; d < 48; ++d) {
    float v = grow[d] * t / (qv * fmaxf(kn[b * CB + h * 48 + d], 1e-12f));
    vals[d] = v;
    mx = fmaxf(mx, v);
  }
  float sum = 0.0f;
  for (int d = 0; d < 48; ++d) { vals[d] = __expf(vals[d] - mx); sum += vals[d]; }
  float inv = 1.0f / sum;
  float* arow = attn + (long)bh * 2304 + c * 48;
  for (int d = 0; d < 48; ++d) arow[d] = vals[d] * inv;
}

// P[b][o][dg] = sum_{j<48} proj[o][h*48+j] * attn[b,h,j,dg%48]  (h = dg/48)
__global__ void pmat_k(const float* __restrict__ proj, const float* __restrict__ attn,
                       __bf16* __restrict__ P) {
  long idx = (long)blockIdx.x * blockDim.x + threadIdx.x;
  if (idx >= (long)NB * CB * CB) return;
  int dg = (int)(idx % CB);
  int o  = (int)((idx / CB) % CB);
  int b  = (int)(idx / (CB * CB));
  int h = dg / 48, d = dg % 48;
  float s = 0.0f;
#pragma unroll 4
  for (int j = 0; j < 48; ++j)
    s += proj[o * CB + h * 48 + j] * attn[(((long)b * 4 + h) * 48 + j) * 48 + d];
  P[idx] = (__bf16)s;
}

// ---------------------------------------------------------------------------
// Host-side orchestration
// ---------------------------------------------------------------------------
static inline size_t alignup(size_t v) { return (v + 255) & ~(size_t)255; }

extern "C" void kernel_launch(void* const* d_in, const int* in_sizes, int n_in,
                              void* d_out, int out_size, void* d_ws, size_t ws_size,
                              hipStream_t stream) {
  (void)in_sizes; (void)n_in; (void)out_size; (void)ws_size;
  const float* x        = (const float*)d_in[0];
  const float* y        = (const float*)d_in[1];
  const float* qk_w     = (const float*)d_in[2];
  const float* qk_dw_w  = (const float*)d_in[3];
  const float* v_w      = (const float*)d_in[4];
  const float* v_dw_w   = (const float*)d_in[5];
  const float* proj_w   = (const float*)d_in[6];
  const float* temper   = (const float*)d_in[7];
  float* out = (float*)d_out;
  char* ws = (char*)d_ws;

  size_t off = 0;
  size_t R0 = off; off += alignup((size_t)NB * CB * NS * 2);      // xb -> yb -> vfull
  size_t R1 = off; off += alignup((size_t)NB * 2 * CB * NS * 2);  // qkb -> vmid
  size_t R2 = off; off += alignup((size_t)NB * CB * NS * 2);      // q
  size_t R3 = off; off += alignup((size_t)NB * CB * NS * 2);      // k
  size_t WQK = off; off += alignup((size_t)2 * CB * CB * 2);
  size_t WV  = off; off += alignup((size_t)CB * CB * 2);
  size_t WDW = off; off += alignup((size_t)CB * 9 * CB * 2);
  size_t PB  = off; off += alignup((size_t)NB * CB * CB * 2);
  size_t GG  = off; off += alignup((size_t)NB * 4 * 48 * 48 * 4);
  size_t ATT = off; off += alignup((size_t)NB * 4 * 48 * 48 * 4);
  size_t QN  = off; off += alignup((size_t)NB * CB * 4);
  size_t KN  = off; off += alignup((size_t)NB * CB * 4);

  __bf16* xb    = (__bf16*)(ws + R0);     // [N][192] pixel-major; reused: yb, vfull
  __bf16* qkb   = (__bf16*)(ws + R1);     // [N][384] pixel-major; reused: vmid [N][192]
  __bf16* qb    = (__bf16*)(ws + R2);     // [N][192]
  __bf16* kb    = (__bf16*)(ws + R3);     // [N][192]
  __bf16* wqk   = (__bf16*)(ws + WQK);
  __bf16* wv    = (__bf16*)(ws + WV);
  __bf16* wdw   = (__bf16*)(ws + WDW);
  __bf16* Pmat  = (__bf16*)(ws + PB);
  float*  G     = (float*)(ws + GG);
  float*  attn  = (float*)(ws + ATT);
  float*  qn    = (float*)(ws + QN);
  float*  kn    = (float*)(ws + KN);

  // 1) transpose-casts and weight packing
  tcast_k<<<dim3(NS / 64, CB / 32, NB), 256, 0, stream>>>(x, xb);
  cast_f32_bf16_k<<<dim3((2 * CB * CB + 255) / 256), 256, 0, stream>>>(qk_w, wqk, 2 * CB * CB);
  cast_f32_bf16_k<<<dim3((CB * CB + 255) / 256), 256, 0, stream>>>(v_w, wv, CB * CB);
  pack_vdw_k<<<dim3(((long)CB * 9 * CB + 255) / 256), 256, 0, stream>>>(v_dw_w, wdw);

  // 2) qk = x @ qk_w^T  -> [N][384] bf16
  gemm_px_bf16<__bf16, false><<<dim3(NS / 128, (2 * CB) / 64, NB), 128, 0, stream>>>(
      wqk, xb, qkb, 2 * CB, CB, 0, 0, (long)CB * NS, (long)2 * CB * NS);

  // 3) depthwise 3x3 -> q, k (pixel-major)
  dwconv3_px_k<<<dim3(NS, NB), 384, 0, stream>>>(qkb, qk_dw_w, qb, kb);

  // 4) vmid = y @ v_w^T  (y cast reuses R0; vmid reuses R1)
  __bf16* yb   = xb;
  __bf16* vmid = qkb;
  tcast_k<<<dim3(NS / 64, CB / 32, NB), 256, 0, stream>>>(y, yb);
  gemm_px_bf16<__bf16, false><<<dim3(NS / 128, CB / 64, NB), 128, 0, stream>>>(
      wv, yb, vmid, CB, CB, 0, 0, (long)CB * NS, (long)CB * NS);

  // 5) vfull = full 3x3 conv (implicit im2col, K=1728) -> R0
  __bf16* vfull = xb;
  conv3_px_bf16<<<dim3(NS / 128, CB / 64, NB), 128, 0, stream>>>(
      wdw, vmid, vfull, (long)CB * NS, (long)CB * NS);

  // 6) norms + Gram + softmax
  zero_f32_k<<<dim3((NB * CB + 255) / 256), 256, 0, stream>>>(qn, NB * CB);
  zero_f32_k<<<dim3((NB * CB + 255) / 256), 256, 0, stream>>>(kn, NB * CB);
  zero_f32_k<<<dim3((NB * 4 * 48 * 48 + 255) / 256), 256, 0, stream>>>(G, NB * 4 * 48 * 48);
  colsq_k<<<dim3(NS / 256, NB), 192, 0, stream>>>(qb, qn);
  colsq_k<<<dim3(NS / 256, NB), 192, 0, stream>>>(kb, kn);
  final_norm_k<<<dim3((NB * CB + 255) / 256), 256, 0, stream>>>(qn, NB * CB);
  final_norm_k<<<dim3((NB * CB + 255) / 256), 256, 0, stream>>>(kn, NB * CB);
  gram48_px<<<dim3(32, NB * 4), 128, 0, stream>>>(qb, kb, G);
  softmax48_k<<<dim3((NB * 4 * 48 + 63) / 64), 64, 0, stream>>>(G, qn, kn, temper, attn);

  // 7) P = proj ∘ blockdiag(attn); out = vfull @ P^T, stored channel-major f32
  pmat_k<<<dim3(((long)NB * CB * CB + 255) / 256), 256, 0, stream>>>(proj_w, attn, Pmat);
  gemm_px_bf16<float, true><<<dim3(NS / 128, CB / 64, NB), 128, 0, stream>>>(
      Pmat, vfull, out, CB, CB, (long)NS, (long)CB * CB, (long)CB * NS, (long)CB * NS);
}
